// ParallelSelfAttention_21663815041324
// MI455X (gfx1250) — matrix operand order repaired and checked by reading, not compiled, not run on hardware
//
#include <hip/hip_runtime.h>
#include <math.h>

// ---------------- problem constants ----------------
#define S_LEN   2048
#define BATCH   2
#define HID     2048
#define NHEAD   16
#define HD      128
#define SB      (S_LEN*BATCH)      // 4096 rows
#define N_QKV   (3*HID)            // 6144
#define INV_NORM 0.08838834764831845f   // 1/sqrt(128)
#define MASK_VAL (-10000.0f)

typedef __attribute__((ext_vector_type(16))) __bf16 v16bf;
typedef __attribute__((ext_vector_type(8)))  float  v8f;

#ifndef __has_builtin
#define __has_builtin(x) 0
#endif
#if __has_builtin(__builtin_amdgcn_tensor_load_to_lds)
#define HAVE_TDM 1
typedef unsigned int v4u  __attribute__((ext_vector_type(4)));
typedef int          v8i_ __attribute__((ext_vector_type(8)));
typedef int          v4i_ __attribute__((ext_vector_type(4)));
#else
#define HAVE_TDM 0
#endif

__device__ inline v8f wmma_bf16(v16bf a, v16bf b, v8f c) {
  return __builtin_amdgcn_wmma_f32_16x16x32_bf16(false, a, false, b, (short)0, c,
                                                 false, false);
}

// Async global->LDS copy of 16 bytes (per-lane), tracked by ASYNCcnt.
__device__ inline void async_ld_b128(const void* lds_ptr, const void* gbase,
                                     unsigned voff) {
  unsigned ldsa = (unsigned)(size_t)lds_ptr;
  asm volatile("global_load_async_to_lds_b128 %0, %1, %2"
               :: "v"(ldsa), "v"(voff), "s"((unsigned long long)(size_t)gbase)
               : "memory");
}
__device__ inline void wait_async0() {
  asm volatile("s_wait_asynccnt 0x0" ::: "memory");
}

// A-fragment (16x32, bf16) from row-major [M][K], row stride `ld`.
__device__ inline v16bf load_a_frag(const __bf16* __restrict__ base, int ld) {
  const int lane = threadIdx.x & 31;
  const int hf = lane >> 4, m = lane & 15;
  const __bf16* p = base + (size_t)m * ld;
  v16bf a;
#pragma unroll
  for (int e = 0; e < 8; ++e) a[e] = p[hf * 8 + e];
#pragma unroll
  for (int e = 0; e < 8; ++e) a[8 + e] = p[16 + hf * 8 + e];
  return a;
}

// B-fragment (32x16, bf16) from N-major [N][K] (pre-transposed), stride `ld`.
__device__ inline v16bf load_bt_frag(const __bf16* __restrict__ base, int ld) {
  const int lane = threadIdx.x & 31;
  const int hf = lane >> 4, n = lane & 15;
  const __bf16* p = base + (size_t)n * ld;
  v16bf b;
#pragma unroll
  for (int e = 0; e < 8; ++e) b[e] = p[hf * 8 + e];
#pragma unroll
  for (int e = 0; e < 8; ++e) b[8 + e] = p[16 + hf * 8 + e];
  return b;
}

// ---------------- kernel 1: fp32 -> bf16 ----------------
__global__ void cvt_f32_bf16(const float* __restrict__ in,
                             __bf16* __restrict__ out, int n4) {
  int i = blockIdx.x * blockDim.x + threadIdx.x;
  if (i < n4) {
    const float4 x = reinterpret_cast<const float4*>(in)[i];
    __bf16* o = out + (size_t)i * 4;
    o[0] = (__bf16)x.x; o[1] = (__bf16)x.y; o[2] = (__bf16)x.z; o[3] = (__bf16)x.w;
  }
}

// ---------------- kernel 2: QKV GEMM (double-buffered) ----------------
#define PADK 40   // 32 + 8 pad -> 80B rows, 16B aligned

__global__ __launch_bounds__(256)
void qkv_gemm(const __bf16* __restrict__ A,   // [4096][2048] bf16
              const __bf16* __restrict__ Bw,  // [2048][6144] bf16
              const float*  __restrict__ bias,// [6144]
              __bf16* __restrict__ Q, __bf16* __restrict__ K,
              __bf16* __restrict__ V) {
  __shared__ __align__(16) __bf16 sA[2][128 * PADK];   // [m][k]
  __shared__ __align__(16) __bf16 sB[2][128 * PADK];   // [n][k] (transposed)
  const int lda = HID, ldb = N_QKV;
  const int m0 = blockIdx.y * 128, n0 = blockIdx.x * 128;
  const int tid = threadIdx.x, lane = tid & 31, wv = tid >> 5;
  const int wm = (wv >> 1) * 32, wn = (wv & 1) * 64;
  const int hf = lane >> 4, lm = lane & 15;

  v8f acc[2][4];
#pragma unroll
  for (int i = 0; i < 2; ++i)
#pragma unroll
    for (int j = 0; j < 4; ++j)
#pragma unroll
      for (int e = 0; e < 8; ++e) acc[i][j][e] = 0.0f;

  const int arow = tid >> 1, aht = tid & 1;            // A stage mapping
  const int bk = tid >> 3, bn0 = (tid & 7) * 16;       // B stage mapping

  auto issueA = [&](int k0, int buf) {
    const unsigned gv =
        (unsigned)((((unsigned)(m0 + arow)) * (unsigned)lda +
                    (unsigned)k0 + (unsigned)aht * 16u) * 2u);
    __bf16* dst = &sA[buf][arow * PADK + aht * 16];
    async_ld_b128(dst,     A, gv);
    async_ld_b128(dst + 8, A, gv + 16);
  };
  auto loadB = [&](int k0, uint4& x0, uint4& x1) {
    const __bf16* gp = Bw + (size_t)(k0 + bk) * ldb + n0 + bn0;
    x0 = *reinterpret_cast<const uint4*>(gp);
    x1 = *reinterpret_cast<const uint4*>(gp + 8);
  };
  auto storeB = [&](int buf, uint4 x0, uint4 x1) {
    __align__(16) __bf16 tmp[16];
    *reinterpret_cast<uint4*>(tmp)     = x0;
    *reinterpret_cast<uint4*>(tmp + 8) = x1;
#pragma unroll
    for (int i = 0; i < 16; ++i) sB[buf][(bn0 + i) * PADK + bk] = tmp[i];
  };

  // prologue: fill buffer 0
  uint4 b0, b1;
  issueA(0, 0);
  loadB(0, b0, b1);
  storeB(0, b0, b1);
  wait_async0();
  __syncthreads();

  for (int k0 = 0, it = 0; k0 < HID; k0 += 32, ++it) {
    const int cur = it & 1, nxt = cur ^ 1;
    const bool hasnext = (k0 + 32) < HID;
    // issue next tile's copies before computing on current tile
    if (hasnext) {
      issueA(k0 + 32, nxt);
      loadB(k0 + 32, b0, b1);
      if (k0 + 64 < HID)
        __builtin_prefetch(Bw + (size_t)(k0 + 64 + bk) * ldb + n0 + bn0, 0, 1);
    }
    // compute on current tile (overlaps async A copy + global B loads)
    v16bf afr[2], bfr[4];
#pragma unroll
    for (int im = 0; im < 2; ++im)
      afr[im] = load_a_frag(&sA[cur][(wm + im * 16) * PADK], PADK);
#pragma unroll
    for (int in = 0; in < 4; ++in)
      bfr[in] = load_bt_frag(&sB[cur][(wn + in * 16) * PADK], PADK);
#pragma unroll
    for (int im = 0; im < 2; ++im)
#pragma unroll
      for (int in = 0; in < 4; ++in)
        acc[im][in] = wmma_bf16(afr[im], bfr[in], acc[im][in]);
    // finish staging next tile
    if (hasnext) storeB(nxt, b0, b1);
    wait_async0();
    __syncthreads();
  }

  // ---- epilogue: scatter into Q/K/V [b][nh][s][hd] as bf16 (+bias) ----
#pragma unroll
  for (int im = 0; im < 2; ++im)
#pragma unroll
    for (int in = 0; in < 4; ++in) {
      const int col = n0 + wn + in * 16 + lm;        // 0..6143
      const int head = col / (3 * HD);
      const int rem = col % (3 * HD);
      const int which = rem >> 7;                    // 0=q 1=k 2=v
      const int d = rem & (HD - 1);
      __bf16* dst = (which == 0) ? Q : (which == 1) ? K : V;
      const float bi = bias[col];
#pragma unroll
      for (int r = 0; r < 8; ++r) {
        const int row = m0 + wm + im * 16 + hf * 8 + r;  // C layout: M=r(+8)
        const int s = row >> 1, b = row & 1;
        dst[(((size_t)b * NHEAD + head) * S_LEN + s) * HD + d] =
            (__bf16)(acc[im][in][r] + bi);
      }
    }
}

// ---------------- kernel 3: flash attention ----------------
#define PADT 72    // 64 + 8 pad (144B rows)
#define PADH 136   // 128 + 8 pad (272B rows)

__global__ __launch_bounds__(256)
void attn_flash(const __bf16* __restrict__ Qg, const __bf16* __restrict__ Kg,
                const __bf16* __restrict__ Vg, __bf16* __restrict__ Ctx) {
  __shared__ __align__(16) __bf16 sVt[128 * PADT];        // [d][t] transposed V
  __shared__ __align__(16) __bf16 sK[64 * PADH];          // [t][d] K block
  __shared__ __align__(16) __bf16 sP[8 * 16 * PADT];      // per-wave P [16][64]

  const int qblk = blockIdx.x & 15;          // S/128 = 16
  const int head = (blockIdx.x >> 4) & 15;
  const int bidx = blockIdx.x >> 8;
  const int tid = threadIdx.x, lane = tid & 31, wv = tid >> 5;
  const int hf = lane >> 4, lm = lane & 15;

  const size_t bh = ((size_t)bidx * NHEAD + head) * S_LEN * HD;
  const __bf16* Qp = Qg + bh;
  const __bf16* Kp = Kg + bh;
  const __bf16* Vp = Vg + bh;

  const int q0 = qblk * 128;
  const int qrow0 = q0 + wv * 16;

  // Q fragments for 4 k-steps of 32 (HD = 128)
  v16bf qf[4];
#pragma unroll
  for (int kk = 0; kk < 4; ++kk)
    qf[kk] = load_a_frag(Qp + (size_t)qrow0 * HD + kk * 32, HD);

  v8f oacc[8];
#pragma unroll
  for (int dt = 0; dt < 8; ++dt)
#pragma unroll
    for (int e = 0; e < 8; ++e) oacc[dt][e] = 0.0f;
  float mrow[8], lrow[8];
#pragma unroll
  for (int r = 0; r < 8; ++r) { mrow[r] = -3.0e38f; lrow[r] = 0.0f; }

  __bf16* sPw = sP + wv * 16 * PADT;
  const int stg_t = tid >> 2, stg_d0 = (tid & 3) * 32;   // staging mapping
  const int nblk = qblk * 2 + 2;                          // causal block count

  for (int jb = 0; jb < nblk; ++jb) {
    const int j0 = jb * 64;
    // ---- stage K block via async global->LDS (layout already Bt) ----
    {
      const unsigned gv =
          (unsigned)((((unsigned)(j0 + stg_t)) * (unsigned)HD +
                      (unsigned)stg_d0) * 2u);
      __bf16* dst = sK + stg_t * PADH + stg_d0;
#pragma unroll
      for (int i = 0; i < 4; ++i)
        async_ld_b128(dst + i * 8, Kp, gv + i * 16u);
    }
    // ---- stage V block transposed (manual): sVt[d][t] = V[j0+t][d] ----
    {
      const __bf16* gp = Vp + (size_t)(j0 + stg_t) * HD + stg_d0;
      __align__(16) __bf16 tmp[32];
      *reinterpret_cast<uint4*>(tmp)      = *reinterpret_cast<const uint4*>(gp);
      *reinterpret_cast<uint4*>(tmp + 8)  = *reinterpret_cast<const uint4*>(gp + 8);
      *reinterpret_cast<uint4*>(tmp + 16) = *reinterpret_cast<const uint4*>(gp + 16);
      *reinterpret_cast<uint4*>(tmp + 24) = *reinterpret_cast<const uint4*>(gp + 24);
#pragma unroll
      for (int i = 0; i < 32; ++i) sVt[(stg_d0 + i) * PADT + stg_t] = tmp[i];
    }
    wait_async0();
    __syncthreads();

    // ---- S = Q . K^T ----
    v8f sf[4];
#pragma unroll
    for (int nt = 0; nt < 4; ++nt)
#pragma unroll
      for (int e = 0; e < 8; ++e) sf[nt][e] = 0.0f;
#pragma unroll
    for (int nt = 0; nt < 4; ++nt)
#pragma unroll
      for (int kk = 0; kk < 4; ++kk) {
        v16bf bf = load_bt_frag(sK + (nt * 16) * PADH + kk * 32, PADH);
        sf[nt] = wmma_bf16(qf[kk], bf, sf[nt]);
      }

    // ---- online softmax per accumulator row ----
#pragma unroll
    for (int r = 0; r < 8; ++r) {
      const int rowM = hf * 8 + r;
      const int sg = qrow0 + rowM;
      float v[4];
      float mx = -3.0e38f;
#pragma unroll
      for (int nt = 0; nt < 4; ++nt) {
        float x = sf[nt][r] * INV_NORM;
        if (j0 + nt * 16 + lm > sg) x = MASK_VAL;
        v[nt] = x;
        mx = fmaxf(mx, x);
      }
#pragma unroll
      for (int i = 1; i < 16; i <<= 1) mx = fmaxf(mx, __shfl_xor(mx, i, 16));
      const float mnew = fmaxf(mrow[r], mx);
      const float corr = __expf(mrow[r] - mnew);
      float ssum = 0.0f;
#pragma unroll
      for (int nt = 0; nt < 4; ++nt) {
        const float p = __expf(v[nt] - mnew);
        ssum += p;
        sPw[rowM * PADT + nt * 16 + lm] = (__bf16)p;
      }
#pragma unroll
      for (int i = 1; i < 16; i <<= 1) ssum += __shfl_xor(ssum, i, 16);
      lrow[r] = lrow[r] * corr + ssum;
      mrow[r] = mnew;
#pragma unroll
      for (int dt = 0; dt < 8; ++dt) oacc[dt][r] *= corr;
    }

    // ---- O += P . V ----
#pragma unroll
    for (int kk = 0; kk < 2; ++kk) {
      v16bf pa = load_a_frag(sPw + kk * 32, PADT);
#pragma unroll
      for (int dt = 0; dt < 8; ++dt) {
        v16bf bv = load_bt_frag(sVt + (dt * 16) * PADT + kk * 32, PADT);
        oacc[dt] = wmma_bf16(pa, bv, oacc[dt]);
      }
    }
    __syncthreads();
  }

  // ---- write context (bf16) rows = s*B+b, cols = head*HD+d ----
#pragma unroll
  for (int dt = 0; dt < 8; ++dt)
#pragma unroll
    for (int r = 0; r < 8; ++r) {
      const int rowM = hf * 8 + r;
      const int sg = qrow0 + rowM;
      const size_t row = (size_t)sg * BATCH + bidx;
      const int col = head * HD + dt * 16 + lm;
      Ctx[row * HID + col] = (__bf16)(oacc[dt][r] / lrow[r]);
    }
}

// ------------- kernel 4: dense projection (TDM, double-buffered) -------------
__global__ __launch_bounds__(256)
void dense_gemm(const __bf16* __restrict__ A,   // [4096][2048] bf16 context
                const __bf16* __restrict__ Bw,  // [2048][2048] bf16
                float* __restrict__ Out) {      // [4096][2048] fp32
  __shared__ __align__(16) __bf16 sA[2][128 * PADK];
  __shared__ __align__(16) __bf16 sB[2][128 * PADK];
  const int lda = HID, ldb = HID;
  const int m0 = blockIdx.y * 128, n0 = blockIdx.x * 128;
  const int tid = threadIdx.x, lane = tid & 31, wv = tid >> 5;
  const int wm = (wv >> 1) * 32, wn = (wv & 1) * 64;
  const int hf = lane >> 4, lm = lane & 15;

  v8f acc[2][4];
#pragma unroll
  for (int i = 0; i < 2; ++i)
#pragma unroll
    for (int j = 0; j < 4; ++j)
#pragma unroll
      for (int e = 0; e < 8; ++e) acc[i][j][e] = 0.0f;

  const int arow = tid >> 1, aht = tid & 1;
  const int bk = tid >> 3, bn0 = (tid & 7) * 16;

  auto issueA = [&](int k0, int buf) {
#if HAVE_TDM
    // Tensor Data Mover: 128x32 bf16 tile, LDS rows padded to 80B via
    // pad_interval=3 (16 DWORDs) / pad_amount=3 (4 DWORDs).  One issue per WG.
    if (wv == 0) {
      const unsigned long long ga =
          (unsigned long long)(size_t)(A + (size_t)m0 * lda + k0);
      const unsigned ldsa = (unsigned)(size_t)&sA[buf][0];
      const v4u g0 = {1u,                                  // count=1
                      ldsa,                                // lds_addr
                      (unsigned)(ga & 0xFFFFFFFFu),        // global_addr lo
                      (unsigned)((ga >> 32) & 0x01FFFFFFu) | (2u << 30)}; // type=2
      const v8i_ g1 = {
          (int)((1u << 16) | (1u << 20) | (3u << 22) | (3u << 25)),
          (int)(((unsigned)HID & 0xFFFFu) << 16),  // tensor_dim0[15:0]=2048
          (int)(((unsigned)SB & 0xFFFFu) << 16),   // tensor_dim1[15:0]=4096
          (int)(32u << 16),                        // tile_dim0 = 32
          128,                                     // tile_dim1 = 128
          (int)HID,                                // tensor_dim0_stride = 2048
          0, 0};
      const v4i_ gz = {0, 0, 0, 0};
#if __has_include(<hip/amd_detail/amd_gfx1250_TDM.h>)
      const v8i_ gz8 = {0, 0, 0, 0, 0, 0, 0, 0};
      __builtin_amdgcn_tensor_load_to_lds(g0, g1, gz, gz, gz8, 0);
#else
      __builtin_amdgcn_tensor_load_to_lds(g0, g1, gz, gz, 0);
#endif
    }
#else
    const unsigned gv =
        (unsigned)((((unsigned)(m0 + arow)) * (unsigned)lda +
                    (unsigned)k0 + (unsigned)aht * 16u) * 2u);
    __bf16* dst = &sA[buf][arow * PADK + aht * 16];
    async_ld_b128(dst,     A, gv);
    async_ld_b128(dst + 8, A, gv + 16);
#endif
  };
  auto waitA = [&]() {
#if HAVE_TDM
    if (wv == 0) __builtin_amdgcn_s_wait_tensorcnt(0);
#else
    wait_async0();
#endif
  };
  auto loadB = [&](int k0, uint4& x0, uint4& x1) {
    const __bf16* gp = Bw + (size_t)(k0 + bk) * ldb + n0 + bn0;
    x0 = *reinterpret_cast<const uint4*>(gp);
    x1 = *reinterpret_cast<const uint4*>(gp + 8);
  };
  auto storeB = [&](int buf, uint4 x0, uint4 x1) {
    __align__(16) __bf16 tmp[16];
    *reinterpret_cast<uint4*>(tmp)     = x0;
    *reinterpret_cast<uint4*>(tmp + 8) = x1;
#pragma unroll
    for (int i = 0; i < 16; ++i) sB[buf][(bn0 + i) * PADK + bk] = tmp[i];
  };

  // prologue: fill buffer 0
  uint4 b0, b1;
  issueA(0, 0);
  loadB(0, b0, b1);
  storeB(0, b0, b1);
  waitA();
  __syncthreads();

  for (int k0 = 0, it = 0; k0 < HID; k0 += 32, ++it) {
    const int cur = it & 1, nxt = cur ^ 1;
    const bool hasnext = (k0 + 32) < HID;
    if (hasnext) {
      issueA(k0 + 32, nxt);
      loadB(k0 + 32, b0, b1);
      if (k0 + 64 < HID)
        __builtin_prefetch(Bw + (size_t)(k0 + 64 + bk) * ldb + n0 + bn0, 0, 1);
    }
    v16bf afr[2], bfr[4];
#pragma unroll
    for (int im = 0; im < 2; ++im)
      afr[im] = load_a_frag(&sA[cur][(wm + im * 16) * PADK], PADK);
#pragma unroll
    for (int in = 0; in < 4; ++in)
      bfr[in] = load_bt_frag(&sB[cur][(wn + in * 16) * PADK], PADK);
#pragma unroll
    for (int im = 0; im < 2; ++im)
#pragma unroll
      for (int in = 0; in < 4; ++in)
        acc[im][in] = wmma_bf16(afr[im], bfr[in], acc[im][in]);
    if (hasnext) storeB(nxt, b0, b1);
    waitA();
    __syncthreads();
  }

#pragma unroll
  for (int im = 0; im < 2; ++im)
#pragma unroll
    for (int in = 0; in < 4; ++in) {
      const int col = n0 + wn + in * 16 + lm;
#pragma unroll
      for (int r = 0; r < 8; ++r) {
        const int row = m0 + wm + im * 16 + hf * 8 + r;
        Out[(size_t)row * HID + col] = acc[im][in][r];
      }
    }
}

// ---------------- kernel 5: bias tail copy ----------------
__global__ void copy_bias(const float* __restrict__ b, float* __restrict__ o, int n) {
  int i = blockIdx.x * blockDim.x + threadIdx.x;
  if (i < n) o[i] = b[i];
}

// ---------------- launch ----------------
extern "C" void kernel_launch(void* const* d_in, const int* in_sizes, int n_in,
                              void* d_out, int out_size, void* d_ws, size_t ws_size,
                              hipStream_t stream) {
  const float* hs     = (const float*)d_in[0];   // [S,B,H] (rows = s*B+b)
  /* d_in[1]: attention_mask (causal, reproduced analytically) */
  const float* wqkv   = (const float*)d_in[2];   // [H, 3H]
  const float* bqkv   = (const float*)d_in[3];   // [3H]
  const float* wdense = (const float*)d_in[4];   // [H, H]
  const float* bdense = (const float*)d_in[5];   // [H]
  float* out = (float*)d_out;

  // workspace layout (bf16), total ~112 MB
  char* ws = (char*)d_ws;
  __bf16* Xbf  = (__bf16*)(ws);
  __bf16* Wqbf = (__bf16*)(ws + (size_t)16777216);
  __bf16* Qbf  = (__bf16*)(ws + (size_t)41943040);
  __bf16* Kbf  = (__bf16*)(ws + (size_t)58720256);
  __bf16* Vbf  = (__bf16*)(ws + (size_t)75497472);
  __bf16* Ctx  = (__bf16*)(ws + (size_t)92274688);
  __bf16* Wdbf = (__bf16*)(ws + (size_t)109051904);

  // 1) fp32 -> bf16 conversions
  {
    int n4 = SB * HID / 4;
    cvt_f32_bf16<<<(n4 + 255) / 256, 256, 0, stream>>>(hs, Xbf, n4);
    n4 = HID * N_QKV / 4;
    cvt_f32_bf16<<<(n4 + 255) / 256, 256, 0, stream>>>(wqkv, Wqbf, n4);
    n4 = HID * HID / 4;
    cvt_f32_bf16<<<(n4 + 255) / 256, 256, 0, stream>>>(wdense, Wdbf, n4);
  }
  // 2) QKV projection (+bias) with scatter into [b][nh][s][hd]
  {
    dim3 grid(N_QKV / 128, SB / 128);   // 48 x 32
    qkv_gemm<<<grid, 256, 0, stream>>>(Xbf, Wqbf, bqkv, Qbf, Kbf, Vbf);
  }
  // 3) causal flash attention -> context bf16
  {
    attn_flash<<<BATCH * NHEAD * (S_LEN / 128), 256, 0, stream>>>(Qbf, Kbf, Vbf, Ctx);
  }
  // 4) dense projection (skip_bias_add) -> fp32 output
  {
    dim3 grid(HID / 128, SB / 128);     // 16 x 32
    dense_gemm<<<grid, 256, 0, stream>>>(Ctx, Wdbf, out);
  }
  // 5) b_dense returned as second tuple element
  copy_bias<<<(HID + 255) / 256, 256, 0, stream>>>(bdense, out + (size_t)SB * HID, HID);
}
